// LSTM_21345987461436
// MI455X (gfx1250) — compile-verified, matrix operand order checked by reading, not run on hardware
//
#include <hip/hip_runtime.h>
#include <hip/hip_bf16.h>

typedef __attribute__((ext_vector_type(16))) __bf16 v16bf;
typedef __attribute__((ext_vector_type(8)))  __bf16 v8bf;
typedef __attribute__((ext_vector_type(8)))  float  v8f;
typedef __attribute__((ext_vector_type(4)))  float  v4f;

#define SSZ 64     // hidden size
#define DD  256    // input dim
#define VV  10000  // vocab
#define BB  16     // batch
#define SL  512    // seq len
#define K4  256    // 4*SS gate width

// ---- WMMA helpers -----------------------------------------------------------
// D = A(16x32 bf16) * B(32x16 bf16) + C(16x16 f32)
__device__ __forceinline__ v8f wmma_bf16(v16bf a, v16bf b, v8f c) {
  return __builtin_amdgcn_wmma_f32_16x16x32_bf16(
      /*neg_a=*/false, a, /*neg_b=*/false, b,
      /*c_mod=*/(short)0, c, /*reuse_a=*/false, /*reuse_b=*/false);
}

// A/B fragment for one 32-wide K window starting at element `koff` of `row`.
// Lane layout (wave32, 16-bit): lanes 0-15 hold K = kb..kb+7 and kb+16..kb+23
// (kb = 8*(lane>>4) folded into koff by the caller) -> two contiguous 16B runs.
__device__ __forceinline__ v16bf frag_from_bf16(const __bf16* row, int koff) {
  v8bf lo = *(const v8bf*)(row + koff);        // K = koff .. koff+7
  v8bf hi = *(const v8bf*)(row + koff + 16);   // K = koff+16 .. koff+23
  v16bf r;
#pragma unroll
  for (int i = 0; i < 8; ++i) { r[i] = lo[i]; r[i + 8] = hi[i]; }
  return r;
}

// Same, but source data is f32 (converted to bf16 on the fly).
__device__ __forceinline__ v16bf frag_from_f32(const float* row, int koff) {
  v4f a0 = *(const v4f*)(row + koff);
  v4f a1 = *(const v4f*)(row + koff + 4);
  v4f b0 = *(const v4f*)(row + koff + 16);
  v4f b1 = *(const v4f*)(row + koff + 20);
  v16bf r;
#pragma unroll
  for (int i = 0; i < 4; ++i) {
    r[i]      = (__bf16)a0[i];
    r[i + 4]  = (__bf16)a1[i];
    r[i + 8]  = (__bf16)b0[i];
    r[i + 12] = (__bf16)b1[i];
  }
  return r;
}

// ---- Phase 0: weights f32 -> bf16 ------------------------------------------
__global__ void k_cvt_bf16(const float* __restrict__ src, __bf16* __restrict__ dst, int n) {
  int i = blockIdx.x * blockDim.x + threadIdx.x;
  if (i < n) dst[i] = (__bf16)src[i];
}

// ---- Phase 1: xproj[s][b][k] = x[b][s][:] . Wxs[k][:] + (Wxs_b[k]+Wss_b[k]) -
// One wave per 16x16 tile; tile M-rows = 16 batches of a fixed timestep s.
// 8192 waves total (512 s-tiles x 16 n-tiles). K = 256 -> 8 bf16 WMMAs.
__global__ void __launch_bounds__(256) k_xproj(const float* __restrict__ x,
                                               const __bf16* __restrict__ WxsB,
                                               const float* __restrict__ bxs,
                                               const float* __restrict__ bss,
                                               float* __restrict__ xproj) {
  const int lane = threadIdx.x & 31;
  const int wid  = blockIdx.x * (blockDim.x >> 5) + (threadIdx.x >> 5);
  const int s    = wid >> 4;            // timestep (M tile)
  const int n0   = (wid & 15) * 16;     // gate-column tile
  const int mn   = lane & 15;           // A row (=batch) / B col index
  const int half = lane >> 4;
  const int kb   = half * 8;

  const float*  arow = x    + ((size_t)mn * SL + s) * DD;   // x[b=mn][s][*]
  const __bf16* brow = WxsB + (size_t)(n0 + mn) * DD;       // Wxs[k=n0+mn][*]

  v8f acc = {};
#pragma unroll
  for (int k0 = 0; k0 < DD; k0 += 32) {
    v16bf a = frag_from_f32 (arow, k0 + kb);
    v16bf b = frag_from_bf16(brow, k0 + kb);
    acc = wmma_bf16(a, b, acc);
  }
  const float bias = bxs[n0 + mn] + bss[n0 + mn];
  float* obase = xproj + (size_t)s * BB * K4 + n0 + mn;
#pragma unroll
  for (int r = 0; r < 8; ++r) {
    int mo = r + half * 8;              // C layout: VGPR r -> row r (+8 upper half)
    obase[(size_t)mo * K4] = acc[r] + bias;
  }
}

// ---- Phase 2: sequential LSTM scan ------------------------------------------
// One block, 16 waves. Wave w computes gate columns [16w,16w+16) via 2 WMMAs
// (Wss fragments register-resident for all 512 steps), then computes gates for
// batch b=w (cell state in registers). xproj C-operand is double-buffered
// across the barrier to hide the global-load latency.
__global__ void __launch_bounds__(512) k_lstm(const float* __restrict__ xproj,
                                              const __bf16* __restrict__ WssB,
                                              __bf16* __restrict__ hs) {
  __shared__ __bf16 hS[BB][80];         // h (bf16), 160B row stride (16B aligned)
  __shared__ float  combS[BB][260];     // gate pre-activations

  const int t    = threadIdx.x;
  const int lane = t & 31;
  const int w    = t >> 5;              // wave id 0..15
  const int n0   = w * 16;
  const int mn   = lane & 15;
  const int half = lane >> 4;
  const int kb   = half * 8;

  for (int i = t; i < BB * 80; i += 512) ((__bf16*)hS)[i] = (__bf16)0.0f;

  // Register-resident B fragments: B[j][n] = Wss[n0+n][j], K windows 0 and 32.
  const __bf16* brow = WssB + (size_t)(n0 + mn) * SSZ;
  const v16bf B0 = frag_from_bf16(brow, 0  + kb);
  const v16bf B1 = frag_from_bf16(brow, 32 + kb);

  // Prefetch xproj C-operand for step 0.
  v8f xp;
  {
    const float* p = xproj + n0 + mn;
#pragma unroll
    for (int r = 0; r < 8; ++r) xp[r] = p[(size_t)(r + half * 8) * K4];
  }

  float c0 = 0.0f, c1 = 0.0f;
  const int gb  = w;                    // batch handled in the gate phase
  const int gj0 = 2 * lane;             // its two hidden columns

  __syncthreads();

  for (int s = 0; s < SL; ++s) {
    // comb tile = xp + h @ Wss^T   (K = 64 -> 2 WMMAs)
    v16bf a0 = frag_from_bf16(&hS[mn][0], 0  + kb);
    v16bf a1 = frag_from_bf16(&hS[mn][0], 32 + kb);
    v8f acc = xp;
    acc = wmma_bf16(a0, B0, acc);
    acc = wmma_bf16(a1, B1, acc);
#pragma unroll
    for (int r = 0; r < 8; ++r) combS[r + half * 8][n0 + mn] = acc[r];

    if (s + 1 < SL) {                   // uniform branch; overlaps with barrier
      const float* p = xproj + (size_t)(s + 1) * BB * K4 + n0 + mn;
#pragma unroll
      for (int r = 0; r < 8; ++r) xp[r] = p[(size_t)(r + half * 8) * K4];
    }
    __syncthreads();

    // gates: f | i | g | o at column offsets 0,64,128,192
    float f0 = combS[gb][gj0],       f1 = combS[gb][gj0 + 1];
    float i0 = combS[gb][64 + gj0],  i1 = combS[gb][64 + gj0 + 1];
    float g0 = combS[gb][128 + gj0], g1 = combS[gb][128 + gj0 + 1];
    float o0 = combS[gb][192 + gj0], o1 = combS[gb][192 + gj0 + 1];
    f0 = 1.0f / (1.0f + __expf(-f0));  f1 = 1.0f / (1.0f + __expf(-f1));
    i0 = 1.0f / (1.0f + __expf(-i0));  i1 = 1.0f / (1.0f + __expf(-i1));
    o0 = 1.0f / (1.0f + __expf(-o0));  o1 = 1.0f / (1.0f + __expf(-o1));
    g0 = tanhf(g0);                    g1 = tanhf(g1);
    c0 = c0 * f0 + i0 * g0;            c1 = c1 * f1 + i1 * g1;
    float h0 = o0 * c0, h1 = o1 * c1;  // reference omits tanh(c) — stay faithful
    __bf16 hb0 = (__bf16)h0, hb1 = (__bf16)h1;
    hS[gb][gj0]     = hb0;
    hS[gb][gj0 + 1] = hb1;
    __bf16* hrow = hs + ((size_t)s * BB + gb) * SSZ + gj0;
    hrow[0] = hb0;
    hrow[1] = hb1;
    __syncthreads();
  }
}

// ---- Phase 3: y[b][s][v] = hs[s][b][:] . Wsy[v][:] + bsy[v] -----------------
// Each wave: one timestep's 16x64 A block (loaded once), 25 N-tiles of 16 vocab
// columns (2 WMMAs each). 512 * 25 = 12800 waves. Store stream (327 MB) is the
// HBM roofline floor (~14 us @ 23.3 TB/s); Wsy bf16 (1.25 MB) lives in L2.
#define NT_PER_WAVE 25
__global__ void __launch_bounds__(256) k_vocab(const __bf16* __restrict__ hs,
                                               const __bf16* __restrict__ WsyB,
                                               const float* __restrict__ bsy,
                                               float* __restrict__ y) {
  const int lane = threadIdx.x & 31;
  const int wid  = blockIdx.x * (blockDim.x >> 5) + (threadIdx.x >> 5);
  const int s      = wid / NT_PER_WAVE;
  const int nchunk = wid % NT_PER_WAVE;
  const int mn   = lane & 15;
  const int half = lane >> 4;
  const int kb   = half * 8;

  const __bf16* arow = hs + ((size_t)s * BB + mn) * SSZ;
  const v16bf a0 = frag_from_bf16(arow, 0  + kb);
  const v16bf a1 = frag_from_bf16(arow, 32 + kb);

  for (int i = 0; i < NT_PER_WAVE; ++i) {
    const int n0 = (nchunk * NT_PER_WAVE + i) * 16;
    const __bf16* brow = WsyB + (size_t)(n0 + mn) * SSZ;
    v16bf b0 = frag_from_bf16(brow, 0  + kb);
    v16bf b1 = frag_from_bf16(brow, 32 + kb);
    v8f acc = {};
    acc = wmma_bf16(a0, b0, acc);
    acc = wmma_bf16(a1, b1, acc);
    const float bias = bsy[n0 + mn];
#pragma unroll
    for (int r = 0; r < 8; ++r) {
      int b = r + half * 8;
      y[((size_t)b * SL + s) * VV + n0 + mn] = acc[r] + bias;
    }
  }
}

// ---- Host-side orchestration -----------------------------------------------
extern "C" void kernel_launch(void* const* d_in, const int* in_sizes, int n_in,
                              void* d_out, int out_size, void* d_ws, size_t ws_size,
                              hipStream_t stream) {
  const float* x     = (const float*)d_in[0];
  const float* Wxs_w = (const float*)d_in[1];
  const float* Wxs_b = (const float*)d_in[2];
  const float* Wss_w = (const float*)d_in[3];
  const float* Wss_b = (const float*)d_in[4];
  const float* Wsy_w = (const float*)d_in[5];
  const float* Wsy_b = (const float*)d_in[6];
  float* y = (float*)d_out;
  (void)in_sizes; (void)n_in; (void)out_size; (void)ws_size;

  // Workspace layout (bytes):
  //   [0,        8388608)  xproj   f32 [512][16][256] (both biases pre-added)
  //   [8388608,  9437184)  hs      bf16 [512][16][64]
  //   [9437184,  9568256)  Wxs     bf16 [256][256]
  //   [9568256,  9601024)  Wss     bf16 [256][64]
  //   [9601024, 10881024)  Wsy     bf16 [10000][64]
  char*   ws    = (char*)d_ws;
  float*  xproj = (float*)(ws);
  __bf16* hsW   = (__bf16*)(ws + 8388608);
  __bf16* WxsB  = (__bf16*)(ws + 9437184);
  __bf16* WssB  = (__bf16*)(ws + 9568256);
  __bf16* WsyB  = (__bf16*)(ws + 9601024);

  k_cvt_bf16<<<(DD * K4 + 255) / 256, 256, 0, stream>>>(Wxs_w, WxsB, DD * K4);
  k_cvt_bf16<<<(K4 * SSZ + 255) / 256, 256, 0, stream>>>(Wss_w, WssB, K4 * SSZ);
  k_cvt_bf16<<<(VV * SSZ + 255) / 256, 256, 0, stream>>>(Wsy_w, WsyB, VV * SSZ);

  // 8192 tiles, 8 waves/block -> 1024 blocks (exact; no divergence near WMMA)
  k_xproj<<<1024, 256, 0, stream>>>(x, WxsB, Wxs_b, Wss_b, xproj);

  // Sequential scan: single workgroup of 16 waves
  k_lstm<<<1, 512, 0, stream>>>(xproj, WssB, hsW);

  // 12800 waves, 8 waves/block -> 1600 blocks (exact)
  k_vocab<<<1600, 256, 0, stream>>>(hsW, WsyB, Wsy_b, y);
}
